// NodeGAT_68925635166336
// MI455X (gfx1250) — matrix operand order repaired and checked by reading, not compiled, not run on hardware
//
#include <hip/hip_runtime.h>
#include <hip/hip_bf16.h>

#define ALPHA 0.2f
#define B_    8
#define N_    2048
#define F_    256
#define OUT_  128
#define CH_   128   // j-chunk width in kernel 2 (4 waves x K=32)

typedef __attribute__((ext_vector_type(16))) __bf16 v16bf;
typedef __attribute__((ext_vector_type(8)))  __bf16 v8bf;
typedef __attribute__((ext_vector_type(8)))  float  v8f;
typedef __attribute__((ext_vector_type(16))) float  v16f;

static __device__ __forceinline__ float lrelu(float x) {
    return x > 0.f ? x : ALPHA * x;
}

// ---------------------------------------------------------------------------
// Kernel 0: pre-convert W (128x256 f32) into bf16, swizzled into the exact
// WMMA A-operand order so kernel 1 does one contiguous 32B load per tile.
// Wsw[(((f0b*8 + t)*2 + half)*16 + m)*16 + e] = bf16(W[16t+m][32*f0b + k(e)])
//   with k(e) = 16*(e>=8) + 8*half + (e&7)
// ---------------------------------------------------------------------------
__global__ __launch_bounds__(256) void gat_wconv(
    const float* __restrict__ W, __bf16* __restrict__ Wsw)
{
  const int id   = blockIdx.x * 256 + threadIdx.x;   // 0..32767
  const int e    = id & 15;
  const int m    = (id >> 4) & 15;
  const int half = (id >> 8) & 1;
  const int t    = (id >> 9) & 7;
  const int f0b  = id >> 12;
  const int k    = ((e & 8) << 1) + 8 * half + (e & 7);
  Wsw[id] = (__bf16)W[(size_t)(16 * t + m) * F_ + 32 * f0b + k];
}

// ---------------------------------------------------------------------------
// Kernel 1:  hT[b][o][n] = sum_f W[o][f] * Nodes[b][n][f]   (stored bf16)
//            Wh1[b][n]   = sum_o hT[o][n] * A[o]
//            Wh2[b][n]   = sum_o hT[o][n] * A[128+o]
// One wave per 16-node tile, full 128-out coverage (8 WMMA col tiles).
// All 8 A-tiles are loaded up-front per k-step so the loads clause together
// and WMMA issue overlaps the load latency.
// ---------------------------------------------------------------------------
__global__ __launch_bounds__(256) void gat_proj(
    const float* __restrict__ Nodes, const __bf16* __restrict__ Wsw,
    const float* __restrict__ A, __bf16* __restrict__ hT,
    float* __restrict__ Wh1, float* __restrict__ Wh2)
{
  const int lane = threadIdx.x & 31;
  const int wave = threadIdx.x >> 5;
  const int tile = blockIdx.x * 8 + wave;       // 0..1023 = 8 batches * 128 tiles
  const int b    = tile >> 7;
  const int n0   = (tile & 127) << 4;
  const int m    = lane & 15;
  const int half = lane >> 4;

  v8f acc[8];
  #pragma unroll
  for (int t = 0; t < 8; ++t) acc[t] = (v8f){0.f,0.f,0.f,0.f,0.f,0.f,0.f,0.f};

  const float* nodep = Nodes + ((size_t)b * N_ + n0 + m) * F_;

  for (int f0b = 0; f0b < 8; ++f0b) {
    // A operands: pre-swizzled bf16 W, 8 x 32B loads issued as one clause
    const __bf16* wp = Wsw + (size_t)(((f0b * 8) * 2 + half) * 16 + m) * 16;
    v16bf amats[8];
    #pragma unroll
    for (int t = 0; t < 8; ++t)
      amats[t] = *(const v16bf*)(wp + (size_t)t * 512);

    // B operand: Nodes[b][n0+m][32*f0b + k(e)], k(e) = 16*(e>=8) + 8*half + (e&7)
    v8f nlo = *(const v8f*)(nodep + 32 * f0b + 8 * half);
    v8f nhi = *(const v8f*)(nodep + 32 * f0b + 16 + 8 * half);
    v16bf bmat;
    #pragma unroll
    for (int e = 0; e < 8; ++e) { bmat[e] = (__bf16)nlo[e]; bmat[8 + e] = (__bf16)nhi[e]; }

    #pragma unroll
    for (int t = 0; t < 8; ++t)
      acc[t] = __builtin_amdgcn_wmma_f32_16x16x32_bf16(
          false, amats[t], false, bmat, (short)0, acc[t], false, false);
  }

  // acc[t][r] = hT[16t + r + 8*half][n0 + m]
  float p1 = 0.f, p2 = 0.f;
  #pragma unroll
  for (int t = 0; t < 8; ++t) {
    #pragma unroll
    for (int r = 0; r < 8; ++r) {
      const int orow = 16 * t + r + 8 * half;
      const float v = acc[t][r];
      hT[((size_t)b * OUT_ + orow) * N_ + n0 + m] = (__bf16)v;
      p1 += v * A[orow];
      p2 += v * A[OUT_ + orow];
    }
  }
  p1 += __shfl_xor(p1, 16);
  p2 += __shfl_xor(p2, 16);
  if (half == 0) {
    Wh1[(size_t)b * N_ + n0 + m] = p1;
    Wh2[(size_t)b * N_ + n0 + m] = p2;
  }
}

// ---------------------------------------------------------------------------
// Kernel 2: fused attention, cooperative flash-style online softmax.
// One block (4 waves, 128 threads) per 16-row i-tile; 4096 waves total.
// Per 128-wide j-chunk: wave w computes softmax for j-slice [32w,32w+32),
// publishes bf16 P + row stats through LDS, then all waves run WMMA over the
// full chunk (K = 4 x 32) on their 2 private out-tiles. All 8 H-tiles and
// 4 P-slices are fetched before the WMMA chain for memory-level parallelism.
// ---------------------------------------------------------------------------
__global__ __launch_bounds__(128) void gat_attn(
    const float* __restrict__ o, const float* __restrict__ Wh1,
    const float* __restrict__ Wh2, const __bf16* __restrict__ hT,
    float* __restrict__ out)
{
  __shared__ __bf16 Pbuf[4][16][40];   // 40 bf16/row: 80B stride, 16B aligned,
                                       // 20-dword bank stride -> conflict-free
  __shared__ float  smax[4][16];
  __shared__ float  ssum[4][16];

  const int lane = threadIdx.x & 31;
  const int w    = threadIdx.x >> 5;            // wave 0..3
  const int b    = blockIdx.x >> 7;
  const int i0   = (blockIdx.x & 127) << 4;
  const int m    = lane & 15;                   // query row within tile
  const int half = lane >> 4;

  const float   wh1  = Wh1[(size_t)b * N_ + i0 + m];
  const float*  orow = o   + ((size_t)b * N_ + i0 + m) * N_;
  const float*  w2p  = Wh2 + (size_t)b * N_;
  const __bf16* hTb  = hT  + (size_t)b * OUT_ * N_;

  float mrow = -3.0e38f;
  float lrow = 0.f;
  v8f acc[2];                                   // out-tiles 2w, 2w+1
  acc[0] = (v8f){0.f,0.f,0.f,0.f,0.f,0.f,0.f,0.f};
  acc[1] = (v8f){0.f,0.f,0.f,0.f,0.f,0.f,0.f,0.f};

  for (int c0 = 0; c0 < N_; c0 += CH_) {
    const int jc = c0 + 32 * w;                 // this wave's 32-wide slice
    if (jc + CH_ < N_)
      __builtin_prefetch(orow + jc + CH_, 0, 1);  // global_prefetch_b8

    // ---- slice logits, layout k(e)=16*(e>=8)+8*half+(e&7); A-operand re-reads P as k=16*half+e
    v8f olo = *(const v8f*)(orow + jc + 8 * half);
    v8f ohi = *(const v8f*)(orow + jc + 16 + 8 * half);
    v8f wlo = *(const v8f*)(w2p  + jc + 8 * half);
    v8f whi = *(const v8f*)(w2p  + jc + 16 + 8 * half);
    float z[16];
    #pragma unroll
    for (int e = 0; e < 8; ++e) {
      z[e]     = lrelu(wh1 + wlo[e]) * olo[e];
      z[8 + e] = lrelu(wh1 + whi[e]) * ohi[e];
    }
    float bmax = z[0];
    #pragma unroll
    for (int e = 1; e < 16; ++e) bmax = fmaxf(bmax, z[e]);
    bmax = fmaxf(bmax, __shfl_xor(bmax, 16));
    if (half == 0) smax[w][m] = bmax;
    __syncthreads();

    // ---- shared chunk max -> identical running stats in all 4 waves
    const float cmax = fmaxf(fmaxf(smax[0][m], smax[1][m]),
                             fmaxf(smax[2][m], smax[3][m]));
    const float mnew = fmaxf(mrow, cmax);
    const float corr = __expf(mrow - mnew);

    float psum = 0.f;
    __bf16 pb[16];
    #pragma unroll
    for (int e = 0; e < 16; ++e) {
      const float p = __expf(z[e] - mnew);
      psum += p;
      pb[e] = (__bf16)p;
    }
    v8bf plo, phi;
    #pragma unroll
    for (int e = 0; e < 8; ++e) { plo[e] = pb[e]; phi[e] = pb[8 + e]; }
    *(v8bf*)&Pbuf[w][m][8 * half]      = plo;   // cols  8h..8h+7
    *(v8bf*)&Pbuf[w][m][16 + 8 * half] = phi;   // cols 16+8h..+7
    psum += __shfl_xor(psum, 16);
    if (half == 0) ssum[w][m] = psum;
    __syncthreads();

    lrow = lrow * corr + (ssum[0][m] + ssum[1][m] + ssum[2][m] + ssum[3][m]);
    mrow = mnew;

    // broadcast per-row corr to D-layout rows (r + 8*half)
    float rowc[8];
    #pragma unroll
    for (int r = 0; r < 8; ++r) rowc[r] = __shfl(corr, r + 8 * half);
    #pragma unroll
    for (int u = 0; u < 2; ++u)
      #pragma unroll
      for (int r = 0; r < 8; ++r) acc[u][r] *= rowc[r];

    // ---- fetch all operands first (clause the 8 global + 8 LDS loads) ----
    v16bf hmats[8];
    #pragma unroll
    for (int s = 0; s < 4; ++s)
      #pragma unroll
      for (int u = 0; u < 2; ++u)
        hmats[2 * s + u] = *(const v16bf*)(hTb
            + (size_t)(16 * (2 * w + u) + m) * N_ + c0 + 32 * s + 16 * half);
    v16bf amats[4];
    #pragma unroll
    for (int s = 0; s < 4; ++s) {
      v8bf alo = *(const v8bf*)&Pbuf[s][m][16 * half];
      v8bf ahi = *(const v8bf*)&Pbuf[s][m][16 * half + 8];
      amats[s] = __builtin_shufflevector(alo, ahi,
          0,1,2,3,4,5,6,7,8,9,10,11,12,13,14,15);
    }

    // ---- P @ H over the full 128-wide chunk (4 K-slices x 2 out-tiles) ----
    #pragma unroll
    for (int s = 0; s < 4; ++s)
      #pragma unroll
      for (int u = 0; u < 2; ++u)
        acc[u] = __builtin_amdgcn_wmma_f32_16x16x32_bf16(
            false, amats[s], false, hmats[2 * s + u], (short)0, acc[u],
            false, false);
    // next chunk's smax write is fenced by the first barrier of the next
    // iteration; P reads above happen before each wave re-enters it.
  }

  float rowl[8];
  #pragma unroll
  for (int r = 0; r < 8; ++r) rowl[r] = __shfl(lrow, r + 8 * half);

  #pragma unroll
  for (int u = 0; u < 2; ++u) {
    const int t = 2 * w + u;
    #pragma unroll
    for (int r = 0; r < 8; ++r) {
      const float v = lrelu(acc[u][r] / rowl[r]);
      out[((size_t)b * N_ + i0 + r + 8 * half) * OUT_ + 16 * t + m] = v;
    }
  }
}

// ---------------------------------------------------------------------------
extern "C" void kernel_launch(void* const* d_in, const int* in_sizes, int n_in,
                              void* d_out, int out_size, void* d_ws, size_t ws_size,
                              hipStream_t stream) {
  const float* Nodes = (const float*)d_in[0];   // [8,2048,256]
  const float* o     = (const float*)d_in[1];   // [8,2048,2048]
  const float* W     = (const float*)d_in[2];   // [128,256]
  const float* A     = (const float*)d_in[3];   // [256,1]
  float* out = (float*)d_out;                   // [8,2048,128]

  char* ws = (char*)d_ws;
  __bf16* hT  = (__bf16*)ws;                                   // 4 MB
  float*  Wh1 = (float*)(ws + (size_t)B_ * OUT_ * N_ * 2);     // 64 KB
  float*  Wh2 = Wh1 + (size_t)B_ * N_;                         // 64 KB
  __bf16* Wsw = (__bf16*)(ws + (size_t)B_ * OUT_ * N_ * 2
                             + (size_t)2 * B_ * N_ * 4);       // 64 KB

  gat_wconv<<<128, 256, 0, stream>>>(W, Wsw);
  gat_proj <<<128, 256, 0, stream>>>(Nodes, Wsw, A, hT, Wh1, Wh2);
  gat_attn <<<1024, 128, 0, stream>>>(o, Wh1, Wh2, hT, out);
}
